// BitMGQA_9423158247768
// MI455X (gfx1250) — 16-bit and 8-bit matrix operand order repaired and checked by reading, compiled, not run on hardware
//
#include <hip/hip_runtime.h>
#include <cstdint>


typedef __attribute__((ext_vector_type(8)))  int     v8i;
typedef __attribute__((ext_vector_type(8)))  float   v8f;
typedef __attribute__((ext_vector_type(16))) __bf16  v16bf;

union BFrag { v16bf v; int4 i[2]; };

#define TOKENS   8192
#define BATCH    4
#define SEQ      2048
#define EMB      1024
#define KVEMB    512
#define HKV      4
#define HD       128

__device__ __forceinline__ unsigned short f2bf(float f) {
    unsigned int u = __float_as_uint(f);
    unsigned int r = (u + 0x7FFFu + ((u >> 16) & 1u)) >> 16;
    return (unsigned short)r;
}

__global__ void wstats_partial(const float* __restrict__ w, int n,
                               float* __restrict__ part) {
    __shared__ float r1[256], r2[256];
    float sa = 0.f, s = 0.f;
    for (int i = blockIdx.x * 256 + threadIdx.x; i < n; i += gridDim.x * 256) {
        float x = w[i]; sa += fabsf(x); s += x;
    }
    r1[threadIdx.x] = sa; r2[threadIdx.x] = s; __syncthreads();
    for (int off = 128; off > 0; off >>= 1) {
        if (threadIdx.x < off) {
            r1[threadIdx.x] += r1[threadIdx.x + off];
            r2[threadIdx.x] += r2[threadIdx.x + off];
        }
        __syncthreads();
    }
    if (threadIdx.x == 0) {
        part[blockIdx.x * 2 + 0] = r1[0];
        part[blockIdx.x * 2 + 1] = r2[0];
    }
}

__global__ void wstats_final(const float* __restrict__ part,
                             float* __restrict__ stats, float inv_n) {
    __shared__ float r1[256], r2[256];
    r1[threadIdx.x] = part[threadIdx.x * 2 + 0];
    r2[threadIdx.x] = part[threadIdx.x * 2 + 1];
    __syncthreads();
    for (int off = 128; off > 0; off >>= 1) {
        if (threadIdx.x < off) {
            r1[threadIdx.x] += r1[threadIdx.x + off];
            r2[threadIdx.x] += r2[threadIdx.x + off];
        }
        __syncthreads();
    }
    if (threadIdx.x == 0) {
        stats[0] = r1[0] * inv_n;
        stats[1] = r2[0] * inv_n;
    }
}

__global__ void wsign_kernel(const float* __restrict__ w,
                             const float* __restrict__ stats,
                             int8_t* __restrict__ out, int n) {
    const float e = stats[1];
    for (int i = blockIdx.x * 256 + threadIdx.x; i < n; i += gridDim.x * 256) {
        float x = w[i];
        out[i] = (int8_t)((x > e) ? 1 : ((x < e) ? -1 : 0));
    }
}

__global__ void act_quant(const float* __restrict__ X, int8_t* __restrict__ Q,
                          float* __restrict__ dq, int K) {
    __shared__ float r1[256], r2[256];
    const int t = blockIdx.x;
    const float* x = X + (size_t)t * K;
    float ss = 0.f, am = 0.f;
    for (int c = threadIdx.x; c < K; c += 256) {
        float v = x[c]; ss += v * v; am = fmaxf(am, fabsf(v));
    }
    r1[threadIdx.x] = ss; r2[threadIdx.x] = am; __syncthreads();
    for (int off = 128; off > 0; off >>= 1) {
        if (threadIdx.x < off) {
            r1[threadIdx.x] += r1[threadIdx.x + off];
            r2[threadIdx.x] = fmaxf(r2[threadIdx.x], r2[threadIdx.x + off]);
        }
        __syncthreads();
    }
    const float rms = rsqrtf(r1[0] / (float)K + 1e-6f);
    const float s   = 127.f / fmaxf(r2[0] * rms, 1e-5f);
    if (threadIdx.x == 0) dq[t] = 1.f / s;
    for (int c = threadIdx.x; c < K; c += 256) {
        float v = rintf(x[c] * rms * s);
        v = fminf(fmaxf(v, -128.f), 127.f);
        Q[(size_t)t * K + c] = (int8_t)v;
    }
}

__global__ __launch_bounds__(256, 1)
void gemm_i8(const int8_t* __restrict__ X, const float* __restrict__ dq,
             const int8_t* __restrict__ Wsgn, const float* __restrict__ stats,
             float* __restrict__ Y, int K, int O) {
    const int lane = threadIdx.x & 31;
    const int wave = threadIdx.x >> 5;
    const int half = lane >> 4;
    const int l16  = lane & 15;
    const int row0 = blockIdx.y * 128 + wave * 16;
    const int col0 = blockIdx.x * 16;

    v8i acc = {0, 0, 0, 0, 0, 0, 0, 0};
    const int8_t* ap = X + (size_t)(row0 + l16) * K + half * 16;
    const int8_t* bp = Wsgn + (size_t)(col0 + l16) * K + half * 16;

    for (int kc = 0; kc < K; kc += 64) {
        int2 a0 = *(const int2*)(ap);
        int2 a1 = *(const int2*)(ap + 8);
        int2 a2 = *(const int2*)(ap + 32);
        int2 a3 = *(const int2*)(ap + 40);
        v8i A = {a0.x, a0.y, a1.x, a1.y, a2.x, a2.y, a3.x, a3.y};
        int4 b0 = *(const int4*)(bp);
        int4 b1 = *(const int4*)(bp + 32);
        v8i Bv = {b0.x, b0.y, b0.z, b0.w, b1.x, b1.y, b1.z, b1.w};
        acc = __builtin_amdgcn_wmma_i32_16x16x64_iu8(true, A, true, Bv, acc,
                                                     false, false);
        ap += 64; bp += 64;
    }

    const float wscale = stats[0];
    const int mbase = half * 8;
#pragma unroll
    for (int r = 0; r < 8; ++r) {
        const int row = row0 + mbase + r;
        Y[(size_t)row * O + col0 + l16] = (float)acc[r] * wscale * dq[row];
    }
}

__global__ void prep_q(const float* __restrict__ Y, unsigned short* __restrict__ qe) {
    unsigned int idx = blockIdx.x * 256 + threadIdx.x;
    if (idx >= (unsigned)BATCH * HKV * SEQ * HD) return;
    int d = idx & 127;
    int n = (idx >> 7) & 2047;
    int h = (idx >> 18) & 3;
    int b = idx >> 20;
    size_t src = ((size_t)(b * SEQ + n)) * EMB;
    float v = Y[src + (2 * h) * HD + d] + Y[src + (2 * h + 1) * HD + d];
    qe[idx] = f2bf(v * (1.f / 128.f));
}

__global__ void prep_k(const float* __restrict__ Y, unsigned short* __restrict__ kb) {
    unsigned int idx = blockIdx.x * 256 + threadIdx.x;
    if (idx >= (unsigned)BATCH * HKV * SEQ * HD) return;
    int d = idx & 127;
    int s = (idx >> 7) & 2047;
    int h = (idx >> 18) & 3;
    int b = idx >> 20;
    kb[idx] = f2bf(Y[(size_t)(b * SEQ + s) * KVEMB + h * HD + d]);
}

__global__ void prep_v(const float* __restrict__ Y, unsigned short* __restrict__ vt) {
    unsigned int idx = blockIdx.x * 256 + threadIdx.x;
    if (idx >= (unsigned)BATCH * HKV * HD * SEQ) return;
    int s = idx & 2047;
    int d = (idx >> 11) & 127;
    int h = (idx >> 18) & 3;
    int b = idx >> 20;
    vt[idx] = f2bf(Y[(size_t)(b * SEQ + s) * KVEMB + h * HD + d]);
}

__global__ __launch_bounds__(256, 1)
void attn_kernel(const unsigned short* __restrict__ qeg,
                 const unsigned short* __restrict__ kbg,
                 const unsigned short* __restrict__ vtg,
                 float* __restrict__ xo) {
    __shared__ __align__(16) unsigned short pbuf[8][16 * 32];
    const int lane = threadIdx.x & 31;
    const int wave = threadIdx.x >> 5;
    const int half = lane >> 4;
    const int l16  = lane & 15;
    const int bh   = blockIdx.y;
    const int b    = bh >> 2;
    const int h    = bh & 3;
    const int qrow0 = blockIdx.x * 128 + wave * 16;

    const unsigned short* q  = qeg + (size_t)bh * SEQ * HD;
    const unsigned short* kk = kbg + (size_t)bh * SEQ * HD;
    const unsigned short* vv = vtg + (size_t)bh * HD * SEQ;

    BFrag Qa[4];
#pragma unroll
    for (int t = 0; t < 4; ++t) {
        const unsigned short* qp = q + (size_t)(qrow0 + l16) * HD + t * 32 + half * 8;
        Qa[t].i[0] = *(const int4*)(qp);
        Qa[t].i[1] = *(const int4*)(qp + 16);
    }

    v8f o[8];
#pragma unroll
    for (int dt = 0; dt < 8; ++dt) o[dt] = (v8f){0, 0, 0, 0, 0, 0, 0, 0};
    float mrow[8], lrow[8];
#pragma unroll
    for (int r = 0; r < 8; ++r) { mrow[r] = -1e30f; lrow[r] = 0.f; }

    for (int kc = 0; kc < SEQ; kc += 32) {
        v8f sc[2];
#pragma unroll
        for (int j = 0; j < 2; ++j) {
            v8f c = (v8f){0, 0, 0, 0, 0, 0, 0, 0};
            const int col = kc + j * 16 + l16;
#pragma unroll
            for (int t = 0; t < 4; ++t) {
                BFrag Bb;
                const unsigned short* kp =
                    kk + (size_t)col * HD + t * 32 + half * 8;
                Bb.i[0] = *(const int4*)(kp);
                Bb.i[1] = *(const int4*)(kp + 16);
                c = __builtin_amdgcn_wmma_f32_16x16x32_bf16(
                        false, Qa[t].v, false, Bb.v, (short)0, c, false, false);
            }
            sc[j] = c;
        }
#pragma unroll
        for (int r = 0; r < 8; ++r) {
            float t0 = fmaxf(sc[0][r], sc[1][r]);
            t0 = fmaxf(t0, __shfl_xor(t0, 1, 16));
            t0 = fmaxf(t0, __shfl_xor(t0, 2, 16));
            t0 = fmaxf(t0, __shfl_xor(t0, 4, 16));
            t0 = fmaxf(t0, __shfl_xor(t0, 8, 16));
            const float nm = fmaxf(mrow[r], t0);
            const float alpha = __expf(mrow[r] - nm);
            mrow[r] = nm;
            const float p0 = __expf(sc[0][r] - nm);
            const float p1 = __expf(sc[1][r] - nm);
            float rs = p0 + p1;
            rs += __shfl_xor(rs, 1, 16);
            rs += __shfl_xor(rs, 2, 16);
            rs += __shfl_xor(rs, 4, 16);
            rs += __shfl_xor(rs, 8, 16);
            lrow[r] = lrow[r] * alpha + rs;
#pragma unroll
            for (int dt = 0; dt < 8; ++dt) o[dt][r] *= alpha;
            const int rr = r + half * 8;
            pbuf[wave][rr * 32 + l16]      = f2bf(p0);
            pbuf[wave][rr * 32 + 16 + l16] = f2bf(p1);
        }
        asm volatile("s_wait_dscnt 0" ::: "memory");
        BFrag Pa;
        const unsigned short* pp = &pbuf[wave][l16 * 32 + half * 8];
        Pa.i[0] = *(const int4*)(pp);
        Pa.i[1] = *(const int4*)(pp + 16);
#pragma unroll
        for (int dt = 0; dt < 8; ++dt) {
            BFrag Vb;
            const unsigned short* vp =
                vv + (size_t)(dt * 16 + l16) * SEQ + kc + half * 8;
            Vb.i[0] = *(const int4*)(vp);
            Vb.i[1] = *(const int4*)(vp + 16);
            o[dt] = __builtin_amdgcn_wmma_f32_16x16x32_bf16(
                        false, Pa.v, false, Vb.v, (short)0, o[dt], false, false);
        }
    }
#pragma unroll
    for (int dt = 0; dt < 8; ++dt) {
#pragma unroll
        for (int r = 0; r < 8; ++r) {
            const int n   = qrow0 + r + half * 8;
            const int tkn = b * SEQ + n;
            const int col = h * HD + dt * 16 + l16;
            xo[(size_t)tkn * KVEMB + col] = o[dt][r] / lrow[r];
        }
    }
}

__global__ void layer_norm(const float* __restrict__ X, const float* __restrict__ g,
                           const float* __restrict__ bta, float* __restrict__ Yo) {
    __shared__ float r1[256], r2[256];
    const int t = blockIdx.x;
    const float* x = X + (size_t)t * KVEMB;
    float s = 0.f, sq = 0.f;
    for (int c = threadIdx.x; c < KVEMB; c += 256) {
        float v = x[c]; s += v; sq += v * v;
    }
    r1[threadIdx.x] = s; r2[threadIdx.x] = sq; __syncthreads();
    for (int off = 128; off > 0; off >>= 1) {
        if (threadIdx.x < off) {
            r1[threadIdx.x] += r1[threadIdx.x + off];
            r2[threadIdx.x] += r2[threadIdx.x + off];
        }
        __syncthreads();
    }
    const float mean = r1[0] * (1.f / KVEMB);
    const float var  = fmaxf(r2[0] * (1.f / KVEMB) - mean * mean, 0.f);
    const float inv  = rsqrtf(var + 1e-5f);
    for (int c = threadIdx.x; c < KVEMB; c += 256)
        Yo[(size_t)t * KVEMB + c] = (x[c] - mean) * inv * g[c] + bta[c];
}

static constexpr size_t O_STATS = 0;
static constexpr size_t O_PART  = 256;
static constexpr size_t O_DQ    = O_PART + 256 * 2 * sizeof(float);
static constexpr size_t O_SGN_Q = O_DQ + (size_t)TOKENS * sizeof(float);
static constexpr size_t O_SGN_K = O_SGN_Q + (size_t)EMB * EMB;
static constexpr size_t O_SGN_V = O_SGN_K + (size_t)KVEMB * EMB;
static constexpr size_t O_SGN_O = O_SGN_V + (size_t)KVEMB * EMB;
static constexpr size_t O_XQ    = O_SGN_O + (size_t)EMB * KVEMB;
static constexpr size_t O_Y     = O_XQ + (size_t)TOKENS * EMB;
static constexpr size_t O_QE    = O_Y + (size_t)TOKENS * EMB * 4;
static constexpr size_t O_KB    = O_QE + (size_t)BATCH * HKV * SEQ * HD * 2;
static constexpr size_t O_VT    = O_KB + (size_t)BATCH * HKV * SEQ * HD * 2;
static constexpr size_t O_XA    = O_VT + (size_t)BATCH * HKV * SEQ * HD * 2;
static constexpr size_t O_XL    = O_XA + (size_t)TOKENS * KVEMB * 4;

extern "C" void kernel_launch(void* const* d_in, const int* in_sizes, int n_in,
                              void* d_out, int out_size, void* d_ws, size_t ws_size,
                              hipStream_t stream) {
    const float* query = (const float*)d_in[0];
    const float* keyi  = (const float*)d_in[1];
    const float* vali  = (const float*)d_in[2];
    const float* gamma = (const float*)d_in[7];
    const float* beta  = (const float*)d_in[8];

    char* ws = (char*)d_ws;
    float*  stats = (float*)(ws + O_STATS);
    float*  part  = (float*)(ws + O_PART);
    float*  dq    = (float*)(ws + O_DQ);
    int8_t* sgn[4] = { (int8_t*)(ws + O_SGN_Q), (int8_t*)(ws + O_SGN_K),
                       (int8_t*)(ws + O_SGN_V), (int8_t*)(ws + O_SGN_O) };
    int8_t* xq = (int8_t*)(ws + O_XQ);
    float*  y  = (float*)(ws + O_Y);
    unsigned short* qe = (unsigned short*)(ws + O_QE);
    unsigned short* kb = (unsigned short*)(ws + O_KB);
    unsigned short* vt = (unsigned short*)(ws + O_VT);
    float* xa = (float*)(ws + O_XA);
    float* xl = (float*)(ws + O_XL);

    for (int wgt = 0; wgt < 4; ++wgt) {
        const float* wptr = (const float*)d_in[3 + wgt];
        const int n = in_sizes[3 + wgt];
        wstats_partial<<<256, 256, 0, stream>>>(wptr, n, part);
        wstats_final<<<1, 256, 0, stream>>>(part, stats + 2 * wgt, 1.f / (float)n);
        wsign_kernel<<<(n + 255) / 256, 256, 0, stream>>>(wptr, stats + 2 * wgt,
                                                          sgn[wgt], n);
    }

    const int prepBlocks = (BATCH * HKV * SEQ * HD + 255) / 256;

    act_quant<<<TOKENS, 256, 0, stream>>>(query, xq, dq, EMB);
    gemm_i8<<<dim3(EMB / 16, TOKENS / 128), 256, 0, stream>>>(
        xq, dq, sgn[0], stats + 0, y, EMB, EMB);
    prep_q<<<prepBlocks, 256, 0, stream>>>(y, qe);

    act_quant<<<TOKENS, 256, 0, stream>>>(keyi, xq, dq, EMB);
    gemm_i8<<<dim3(KVEMB / 16, TOKENS / 128), 256, 0, stream>>>(
        xq, dq, sgn[1], stats + 2, y, EMB, KVEMB);
    prep_k<<<prepBlocks, 256, 0, stream>>>(y, kb);

    act_quant<<<TOKENS, 256, 0, stream>>>(vali, xq, dq, EMB);
    gemm_i8<<<dim3(KVEMB / 16, TOKENS / 128), 256, 0, stream>>>(
        xq, dq, sgn[2], stats + 4, y, EMB, KVEMB);
    prep_v<<<prepBlocks, 256, 0, stream>>>(y, vt);

    attn_kernel<<<dim3(SEQ / 128, BATCH * HKV), 256, 0, stream>>>(qe, kb, vt, xa);

    layer_norm<<<TOKENS, 256, 0, stream>>>(xa, gamma, beta, xl);
    act_quant<<<TOKENS, 256, 0, stream>>>(xl, xq, dq, KVEMB);
    gemm_i8<<<dim3(EMB / 16, TOKENS / 128), 256, 0, stream>>>(
        xq, dq, sgn[3], stats + 6, (float*)d_out, KVEMB, EMB);
}
